// mLSTMCell_42915313222192
// MI455X (gfx1250) — matrix operand order repaired and checked by reading, not compiled, not run on hardware
//
#include <hip/hip_runtime.h>
#include <hip/hip_bf16.h>
#include <math.h>

// ---------------- problem constants ----------------
#define B_   2
#define S_   2048
#define H_   1024
#define NH_  4
#define DH_  256
#define BM_  64          // query rows per workgroup (4 waves x 16)
#define BK_  64          // key chunk (4 x 16-key tiles, 2 PV sub-chunks)
#define NWAVE 4

typedef __attribute__((ext_vector_type(16))) __bf16 v16bf;
typedef __attribute__((ext_vector_type(8)))  float  v8f;

union AFrag {
    v16bf v;
    unsigned short h[16];
    unsigned int   u[8];
};

// native bf16 converts -> v_cvt_pk_bf16_f32 / v_cvt_bf16_f32
__device__ __forceinline__ unsigned int pack2(float a, float b) {
    union { __bf16 h[2]; unsigned int u; } r;
    r.h[0] = (__bf16)a;
    r.h[1] = (__bf16)b;
    return r.u;
}
__device__ __forceinline__ unsigned short bf1(float a) {
    union { __bf16 h; unsigned short u; } r;
    r.h = (__bf16)a;
    return r.u;
}
__device__ __forceinline__ void ld_frag(AFrag& f, const unsigned int* p) {
    uint4 a = *(const uint4*)p;         // -> ds_load_b128
    uint4 b = *(const uint4*)(p + 4);   // -> ds_load_b128
    f.u[0] = a.x; f.u[1] = a.y; f.u[2] = a.z; f.u[3] = a.w;
    f.u[4] = b.x; f.u[5] = b.y; f.u[6] = b.z; f.u[7] = b.w;
}

// ---------------------------------------------------------------------------
// Kernel 1: gate preactivations.  ig = <gin, w_i[h]> + b_i[h]
//           logf = log_sigmoid(<gin, w_f[h]> + b_f[h])
// ---------------------------------------------------------------------------
__global__ __launch_bounds__(128) void gates_kernel(
    const float* __restrict__ q, const float* __restrict__ k, const float* __restrict__ v,
    const float* __restrict__ wi, const float* __restrict__ bi,
    const float* __restrict__ wf, const float* __restrict__ bf,
    float* __restrict__ ig_out, float* __restrict__ logf_out)
{
    int bs   = blockIdx.x;            // 0..B*S-1
    int b    = bs / S_;
    int s    = bs % S_;
    int head = threadIdx.x >> 5;      // wave id == head (NH_==4)
    int lane = threadIdx.x & 31;

    const float* qr = q + ((size_t)(b * S_ + s)) * H_;
    const float* kr = k + ((size_t)(b * S_ + s)) * H_;
    const float* vr = v + ((size_t)(b * S_ + s)) * H_;
    const float* wih = wi + (size_t)head * (3 * H_);
    const float* wfh = wf + (size_t)head * (3 * H_);

    float ai = 0.f, af = 0.f;
    for (int d = lane; d < H_; d += 32) {
        float gq = qr[d], gk = kr[d], gv = vr[d];
        ai += gq * wih[d] + gk * wih[H_ + d] + gv * wih[2 * H_ + d];
        af += gq * wfh[d] + gk * wfh[H_ + d] + gv * wfh[2 * H_ + d];
    }
    #pragma unroll
    for (int m = 16; m; m >>= 1) {
        ai += __shfl_xor(ai, m, 32);
        af += __shfl_xor(af, m, 32);
    }
    if (lane == 0) {
        float igv = ai + bi[head];
        float fgv = af + bf[head];
        float lf = fminf(fgv, 0.f) - log1pf(expf(-fabsf(fgv)));  // stable log-sigmoid
        size_t o = ((size_t)(b * NH_ + head)) * S_ + s;
        ig_out[o]   = igv;
        logf_out[o] = lf;
    }
}

// ---------------------------------------------------------------------------
// Kernel 2: inclusive cumsum of logf along S, one wave per (b,head).
// ---------------------------------------------------------------------------
__global__ __launch_bounds__(32) void scan_kernel(
    const float* __restrict__ logf, float* __restrict__ cf)
{
    int bh   = blockIdx.x;
    int lane = threadIdx.x;
    const float* in  = logf + (size_t)bh * S_;
    float*       out = cf   + (size_t)bh * S_;

    float carry = 0.f;
    for (int c = 0; c < S_; c += 32) {
        float x = in[c + lane];
        #pragma unroll
        for (int d = 1; d < 32; d <<= 1) {
            float y = __shfl_up(x, d, 32);
            if (lane >= d) x += y;
        }
        x += carry;
        out[c + lane] = x;
        carry = __shfl(x, 31, 32);
    }
}

// ---------------------------------------------------------------------------
// Kernel 3: flash-style mLSTM + per-head group-norm epilogue.
// 64-key chunks: 64 WMMAs per chunk, one accumulator rescale per chunk.
// All LDS buffers hold data pre-swizzled into exact WMMA-fragment dword order.
// ---------------------------------------------------------------------------
__global__ __launch_bounds__(128) void mlstm_kernel(
    const float* __restrict__ q, const float* __restrict__ k, const float* __restrict__ v,
    const float* __restrict__ ig, const float* __restrict__ cf,
    const float* __restrict__ lnw, float* __restrict__ out)
{
    // Q: [wave][c(8)][lane][8 dwords] = 32KB  (A-fragments, persistent, pre-scaled by 1/16)
    __shared__ __align__(16) unsigned int Qst[NWAVE * 8 * 32 * 8];
    // K: [(c*4+tile)(32)][lane][8 dwords] = 32KB (B-fragments for QK^T, 4 key tiles)
    __shared__ __align__(16) unsigned int KstP[32 * 32 * 8];
    // V: [(sub*16+j)(32)][lane][8 dwords] = 32KB (B-fragments for PV, 2 sub-chunks)
    __shared__ __align__(16) unsigned int VstP[32 * 32 * 8];
    // P staging: per-wave, two A-fragments (sub-chunks of 32 keys) = 8KB
    __shared__ __align__(16) unsigned int SstP[NWAVE][2][32 * 8];

    const int bh   = blockIdx.y;
    const int b    = bh / NH_;
    const int head = bh % NH_;
    const int m0   = blockIdx.x * BM_;
    const int tid  = threadIdx.x;
    const int wave = tid >> 5;
    const int lane = tid & 31;
    const int half = (lane >> 4) & 1;
    const int l4   = lane & 15;
    const int qm   = m0 + wave * 16;

    const float* cfb = cf + (size_t)bh * S_;
    const float* igb = ig + (size_t)bh * S_;

    // ---- stage Q A-fragments (bf16, frag-packed, scaled by 1/sqrt(DH)=2^-4) ----
    {
        const float* qrow = q + ((size_t)(b * S_ + qm + l4)) * H_ + head * DH_;
        const float sc = 0.0625f;          // exact exponent shift in bf16
        #pragma unroll
        for (int c = 0; c < 8; ++c) {
            int d0 = c * 32 + half * 8;        // K 0..7 (half0) / 8..15 (half1)
            int d1 = c * 32 + 16 + half * 8;   // K 16..23       / 24..31
            unsigned int u[8];
            #pragma unroll
            for (int e = 0; e < 4; ++e) {
                u[e]     = pack2(qrow[d0 + 2 * e] * sc, qrow[d0 + 2 * e + 1] * sc);
                u[4 + e] = pack2(qrow[d1 + 2 * e] * sc, qrow[d1 + 2 * e + 1] * sc);
            }
            unsigned off = (unsigned)(((wave * 8 + c) * 32 + lane) * 8);
            *(uint4*)&Qst[off]     = make_uint4(u[0], u[1], u[2], u[3]);
            *(uint4*)&Qst[off + 4] = make_uint4(u[4], u[5], u[6], u[7]);
        }
    }

    // ---- per-row online state (rows r = i + 8*half) ----
    float mrow[8], lrow[8], cfr[8];
    v8f acc[16];
    #pragma unroll
    for (int j = 0; j < 16; ++j) {
        #pragma unroll
        for (int e = 0; e < 8; ++e) acc[j][e] = 0.f;
    }
    #pragma unroll
    for (int i = 0; i < 8; ++i) {
        mrow[i] = -1e30f;
        lrow[i] = 0.f;
        cfr[i]  = cfb[qm + i + 8 * half];
    }

    const int kend = m0 + BM_;              // causal bound for this block
    for (int kt = 0; kt < kend; kt += BK_) {
        // ---- cooperative load: 64-key K and V chunk -> bf16, frag-packed LDS ----
        #pragma unroll
        for (int it = 0; it < 64; ++it) {
            int p = tid + it * 128;
            // K element pair: key r (0..63), features d,d+1
            int r = p >> 7;
            int d = (p & 127) * 2;
            float2 kv = *(const float2*)&k[((size_t)(b * S_ + kt + r)) * H_ + head * DH_ + d];
            unsigned ko = (unsigned)(((((d >> 5) * 4 + (r >> 4)) * 32) +
                          ((r & 15) + 16 * ((d >> 3) & 1))) * 8 + ((d & 7) >> 1) + 4 * ((d >> 4) & 1));
            KstP[ko] = pack2(kv.x, kv.y);
            // V element pair: keys t,t+1 (0..63), feature dcol
            int dcol = p & 255;
            int t    = (p >> 8) * 2;
            int ts   = t & 31;              // within sub-chunk
            float va = v[((size_t)(b * S_ + kt + t))     * H_ + head * DH_ + dcol];
            float vb = v[((size_t)(b * S_ + kt + t + 1)) * H_ + head * DH_ + dcol];
            unsigned vo = (unsigned)(((((t >> 5) * 16 + (dcol >> 4)) * 32) +
                          ((dcol & 15) + 16 * ((ts >> 3) & 1))) * 8 + ((ts & 7) >> 1) + 4 * ((ts >> 4) & 1));
            VstP[vo] = pack2(va, vb);
        }
        // prefetch next chunk while this one is consumed
        int ktn = kt + BK_;
        if (ktn < kend) {
            #pragma unroll
            for (int i = 0; i < 4; ++i) {
                int idx = tid + i * 128;            // 512 cache lines per operand
                int r   = idx >> 3;
                int off = (idx & 7) * 32;           // 128B granularity
                __builtin_prefetch(&k[((size_t)(b * S_ + ktn + r)) * H_ + head * DH_ + off], 0, 3);
                __builtin_prefetch(&v[((size_t)(b * S_ + ktn + r)) * H_ + head * DH_ + off], 0, 3);
            }
        }
        __syncthreads();

        // ---- S = Q K^T over DH (four 16x16 key tiles) ----
        v8f s[4];
        #pragma unroll
        for (int t = 0; t < 4; ++t) {
            #pragma unroll
            for (int e = 0; e < 8; ++e) s[t][e] = 0.f;
        }
        #pragma unroll
        for (int c = 0; c < 8; ++c) {
            AFrag aq;
            ld_frag(aq, &Qst[(unsigned)(((wave * 8 + c) * 32 + lane) * 8)]);
            #pragma unroll
            for (int t = 0; t < 4; ++t) {
                AFrag bk;
                ld_frag(bk, &KstP[(unsigned)(((c * 4 + t) * 32 + lane) * 8)]);
                s[t] = __builtin_amdgcn_wmma_f32_16x16x32_bf16(
                          false, aq.v, false, bk.v, (short)0, s[t], false, false);
            }
        }

        // ---- gating + online max rescale + signed row sum (once per 64 keys) ----
        float cft[4], igt[4];
        #pragma unroll
        for (int t = 0; t < 4; ++t) {
            cft[t] = cfb[kt + t * 16 + l4];
            igt[t] = igb[kt + t * 16 + l4];
        }
        float alpha[8], cw[4][8];
        #pragma unroll
        for (int i = 0; i < 8; ++i) {
            int srow = qm + i + 8 * half;
            float w[4];
            #pragma unroll
            for (int t = 0; t < 4; ++t) {
                int tc = kt + t * 16 + l4;
                w[t] = (tc <= srow) ? (cfr[i] - cft[t] + igt[t]) : -1e30f;
            }
            float mx = fmaxf(fmaxf(w[0], w[1]), fmaxf(w[2], w[3]));
            #pragma unroll
            for (int msk = 8; msk; msk >>= 1)
                mx = fmaxf(mx, __shfl_xor(mx, msk, 32));
            float mnew = fmaxf(mrow[i], mx);
            float a    = __expf(mrow[i] - mnew);   // 1 if both sentinel
            mrow[i]  = mnew;
            alpha[i] = a;
            float rs = 0.f;
            #pragma unroll
            for (int t = 0; t < 4; ++t) {
                float dd = (w[t] > -1e29f) ? __expf(w[t] - mnew) : 0.f;
                float cx = s[t][i] * dd;         // 1/sqrt(DH) folded into Q
                cw[t][i] = cx;
                rs += cx;
            }
            #pragma unroll
            for (int msk = 8; msk; msk >>= 1)
                rs += __shfl_xor(rs, msk, 32);
            lrow[i] = lrow[i] * a + rs;
        }
        #pragma unroll
        for (int j = 0; j < 16; ++j) {
            #pragma unroll
            for (int i = 0; i < 8; ++i) acc[j][i] *= alpha[i];
        }

        // ---- stage weighted P (C-layout) -> two A-frag-packed sub-chunks ----
        #pragma unroll
        for (int t = 0; t < 4; ++t) {
            unsigned short* sw = (unsigned short*)&SstP[wave][t >> 1][0];
            int tc = (t & 1) * 16 + l4;          // key index within 32-key sub-chunk
            int dl16 = 16 * ((tc >> 3) & 1);
            int e    = ((tc & 7) >> 1) + 4 * ((tc >> 4) & 1);
            int hw   = tc & 1;
            #pragma unroll
            for (int i = 0; i < 8; ++i) {
                int row = i + 8 * half;
                sw[((row + dl16) * 8 + e) * 2 + hw] = bf1(cw[t][i]);
            }
        }
        asm volatile("s_wait_dscnt 0" ::: "memory");  // wave-internal RAW on LDS

        // ---- acc += P V  (two sub-chunks x 16 column tiles) ----
        #pragma unroll
        for (int sub = 0; sub < 2; ++sub) {
            AFrag paf;
            ld_frag(paf, &SstP[wave][sub][lane * 8]);
            #pragma unroll
            for (int j = 0; j < 16; ++j) {
                AFrag bv;
                ld_frag(bv, &VstP[(unsigned)(((sub * 16 + j) * 32 + lane) * 8)]);
                acc[j] = __builtin_amdgcn_wmma_f32_16x16x32_bf16(
                            false, paf.v, false, bv.v, (short)0, acc[j], false, false);
            }
        }
        __syncthreads();
    }

    // ---- epilogue: normalizer + per-head group-norm, store f32 ----
    #pragma unroll
    for (int i = 0; i < 8; ++i) {
        float nrm = fmaxf(fabsf(lrow[i]), __expf(-mrow[i])) + 1e-6f;
        float inv = 1.f / nrm;
        float sum = 0.f;
        #pragma unroll
        for (int j = 0; j < 16; ++j) { float x = acc[j][i] * inv; acc[j][i] = x; sum += x; }
        #pragma unroll
        for (int msk = 8; msk; msk >>= 1) sum += __shfl_xor(sum, msk, 32);
        float mean = sum * (1.f / 256.f);
        float ssq = 0.f;
        #pragma unroll
        for (int j = 0; j < 16; ++j) { float d = acc[j][i] - mean; ssq += d * d; }
        #pragma unroll
        for (int msk = 8; msk; msk >>= 1) ssq += __shfl_xor(ssq, msk, 32);
        float rstd = rsqrtf(ssq * (1.f / 256.f) + 1e-5f);

        int srow = qm + i + 8 * half;
        float* orow = out + ((size_t)(b * S_ + srow)) * H_ + head * DH_;
        #pragma unroll
        for (int j = 0; j < 16; ++j)
            orow[j * 16 + l4] = (acc[j][i] - mean) * rstd *
                                (1.f + lnw[head * DH_ + j * 16 + l4]);
    }
}

// ---------------------------------------------------------------------------
extern "C" void kernel_launch(void* const* d_in, const int* in_sizes, int n_in,
                              void* d_out, int out_size, void* d_ws, size_t ws_size,
                              hipStream_t stream)
{
    const float* q   = (const float*)d_in[0];
    const float* k   = (const float*)d_in[1];
    const float* v   = (const float*)d_in[2];
    const float* wi  = (const float*)d_in[3];
    const float* bi  = (const float*)d_in[4];
    const float* wf  = (const float*)d_in[5];
    const float* bf  = (const float*)d_in[6];
    const float* lnw = (const float*)d_in[7];
    float* out = (float*)d_out;

    const size_t N = (size_t)B_ * NH_ * S_;
    float* ws    = (float*)d_ws;
    float* igb   = ws;            // B*NH*S
    float* logfb = ws + N;        // B*NH*S
    float* cfb   = ws + 2 * N;    // B*NH*S

    gates_kernel<<<B_ * S_, 128, 0, stream>>>(q, k, v, wi, bi, wf, bf, igb, logfb);
    scan_kernel<<<B_ * NH_, 32, 0, stream>>>(logfb, cfb);

    dim3 grid(S_ / BM_, B_ * NH_);
    mlstm_kernel<<<grid, 128, 0, stream>>>(q, k, v, igb, cfb, lnw, out);
}
